// Block_6485400617579
// MI455X (gfx1250) — matrix operand order repaired and checked by reading, not compiled, not run on hardware
//
#include <hip/hip_runtime.h>
#include <hip/hip_bf16.h>

typedef __attribute__((ext_vector_type(16))) _Float16 v16h;
typedef __attribute__((ext_vector_type(8)))  _Float16 v8h;
typedef __attribute__((ext_vector_type(8)))  float    v8f;

#define TSEQ   2048
#define NB     2
#define NHEAD  16
#define HDIM   64
#define CEMB   1024
#define NHID   2816
#define MROWS  (NB * TSEQ)   // 4096
#define NEGBIG (-3.0e38f)

// ---------------------------------------------------------------------------
// Weight cast + transpose:  in f32 [K,N] row-major  ->  out f16 [N,K] row-major
// ---------------------------------------------------------------------------
__global__ void cast_wt_kernel(const float* __restrict__ in, _Float16* __restrict__ out,
                               int K, int N) {
  size_t idx = (size_t)blockIdx.x * blockDim.x + threadIdx.x;
  size_t total = (size_t)K * N;
  if (idx >= total) return;
  int k = (int)(idx / N);
  int n = (int)(idx % N);
  out[(size_t)n * K + k] = (_Float16)in[idx];
}

// ---------------------------------------------------------------------------
// RMSNorm + cast to f16.  One row (C=1024) per block of 256 threads.
// ---------------------------------------------------------------------------
__global__ void rmsnorm_cast_kernel(const float* __restrict__ x, const float* __restrict__ scale,
                                    _Float16* __restrict__ out, int C) {
  __shared__ float red[256];
  int row = blockIdx.x;
  const float* xr = x + (size_t)row * C;
  float s = 0.f;
  for (int i = threadIdx.x; i < C; i += 256) { float v = xr[i]; s += v * v; }
  red[threadIdx.x] = s;
  __syncthreads();
  for (int off = 128; off > 0; off >>= 1) {
    if (threadIdx.x < off) red[threadIdx.x] += red[threadIdx.x + off];
    __syncthreads();
  }
  float r = rsqrtf(red[0] / (float)C + 1e-5f);
  for (int i = threadIdx.x; i < C; i += 256)
    out[(size_t)row * C + i] = (_Float16)(xr[i] * r * scale[i]);
}

// ---------------------------------------------------------------------------
// Generic WMMA GEMM:  C[M,N] = A[M,K] (f16, row-major) x Bt[N,K] (f16, row-major = B^T)
// One wave computes a 16(M) x 64(N) tile.  Block = 256 threads = 8 waves -> 128 M rows.
// mode 0: outF32 = acc
// mode 1: outF32 = acc + res
// mode 2: outF16 = silu(res) * acc      (res = fc1 pre-activation, acc = fc2)
// ---------------------------------------------------------------------------
__global__ void __launch_bounds__(256)
gemm_wmma_kernel(const _Float16* __restrict__ A, const _Float16* __restrict__ Bt,
                 float* __restrict__ outF32, _Float16* __restrict__ outF16,
                 const float* __restrict__ res,
                 int M, int N, int K, int mode) {
  int wave = threadIdx.x >> 5;
  int lane = threadIdx.x & 31;
  int hi   = lane >> 4;        // lane half (0/1)
  int l16  = lane & 15;
  int kbA  = hi * 16;          // A-fragment K sub-offset (contiguous 16)
  int kbB  = hi * 16;          // B-fragment K sub-offset (contiguous 16)
  int m0 = (blockIdx.y * 8 + wave) * 16;
  int n0 = blockIdx.x * 64;
  if (m0 >= M || n0 >= N) return;

  v8f acc0 = {}, acc1 = {}, acc2 = {}, acc3 = {};
  const _Float16* arow = A + (size_t)(m0 + l16) * K;

  for (int k0 = 0; k0 < K; k0 += 32) {
    union { v16h v; v8h h[2]; } ua;
    const _Float16* ap = arow + k0 + kbA;
    ua.h[0] = *(const v8h*)(ap);
    ua.h[1] = *(const v8h*)(ap + 8);
    __builtin_prefetch(ap + 64, 0, 1);   // global_prefetch_b8 ahead in K

    union { v16h v; v8h h[2]; } ub0, ub1, ub2, ub3;
    const _Float16* bp0 = Bt + (size_t)(n0 +       l16) * K + k0 + kbB;
    const _Float16* bp1 = Bt + (size_t)(n0 + 16 + l16) * K + k0 + kbB;
    const _Float16* bp2 = Bt + (size_t)(n0 + 32 + l16) * K + k0 + kbB;
    const _Float16* bp3 = Bt + (size_t)(n0 + 48 + l16) * K + k0 + kbB;
    ub0.h[0] = *(const v8h*)(bp0); ub0.h[1] = *(const v8h*)(bp0 + 8);
    ub1.h[0] = *(const v8h*)(bp1); ub1.h[1] = *(const v8h*)(bp1 + 8);
    ub2.h[0] = *(const v8h*)(bp2); ub2.h[1] = *(const v8h*)(bp2 + 8);
    ub3.h[0] = *(const v8h*)(bp3); ub3.h[1] = *(const v8h*)(bp3 + 8);

    acc0 = __builtin_amdgcn_wmma_f32_16x16x32_f16(false, ua.v, false, ub0.v, (short)0, acc0, false, false);
    acc1 = __builtin_amdgcn_wmma_f32_16x16x32_f16(false, ua.v, false, ub1.v, (short)0, acc1, false, false);
    acc2 = __builtin_amdgcn_wmma_f32_16x16x32_f16(false, ua.v, false, ub2.v, (short)0, acc2, false, false);
    acc3 = __builtin_amdgcn_wmma_f32_16x16x32_f16(false, ua.v, false, ub3.v, (short)0, acc3, false, false);
  }

  v8f accs[4] = {acc0, acc1, acc2, acc3};
#pragma unroll
  for (int nt = 0; nt < 4; ++nt) {
    int col = n0 + nt * 16 + l16;
#pragma unroll
    for (int r = 0; r < 8; ++r) {
      int row = m0 + hi * 8 + r;
      size_t idx = (size_t)row * N + col;
      float v = accs[nt][r];
      if (mode == 0) {
        outF32[idx] = v;
      } else if (mode == 1) {
        outF32[idx] = v + res[idx];
      } else {
        float u = res[idx];
        float sil = u / (1.f + __expf(-u));
        outF16[idx] = (_Float16)(sil * v);
      }
    }
  }
}

// ---------------------------------------------------------------------------
// RoPE + repack:  qkv f32 [B,T,3C]  ->  q,k f16 [B,H,T,64],  v f16 transposed [B,H,64,T]
// One thread per (b,h,t,pair).
// ---------------------------------------------------------------------------
__global__ void rope_pack_kernel(const float* __restrict__ qkv,
                                 _Float16* __restrict__ qf, _Float16* __restrict__ kf,
                                 _Float16* __restrict__ vtf) {
  size_t idx = (size_t)blockIdx.x * blockDim.x + threadIdx.x;
  size_t total = (size_t)NB * NHEAD * TSEQ * (HDIM / 2);
  if (idx >= total) return;
  int i  = (int)(idx % (HDIM / 2));          // rotation pair index
  int t  = (int)((idx / (HDIM / 2)) % TSEQ);
  int h  = (int)((idx / ((size_t)(HDIM / 2) * TSEQ)) % NHEAD);
  int b  = (int)(idx / ((size_t)(HDIM / 2) * TSEQ * NHEAD));
  int bh = b * NHEAD + h;

  const float* base = qkv + ((size_t)(b * TSEQ + t)) * (3 * CEMB) + h * HDIM + 2 * i;
  float qe = base[0],            qo = base[1];
  float ke = base[CEMB],         ko = base[CEMB + 1];
  float ve = base[2 * CEMB],     vo = base[2 * CEMB + 1];

  float theta = __powf(10000.f, -(float)(2 * i) / (float)HDIM);
  float sn, cs;
  __sincosf((float)t * theta, &sn, &cs);

  size_t qki = ((size_t)bh * TSEQ + t) * HDIM + 2 * i;
  qf[qki]     = (_Float16)(qe * cs - qo * sn);
  qf[qki + 1] = (_Float16)(qe * sn + qo * cs);
  kf[qki]     = (_Float16)(ke * cs - ko * sn);
  kf[qki + 1] = (_Float16)(ke * sn + ko * cs);
  // v transposed: [B,H,64,T]
  vtf[((size_t)bh * HDIM + 2 * i)     * TSEQ + t] = (_Float16)ve;
  vtf[((size_t)bh * HDIM + 2 * i + 1) * TSEQ + t] = (_Float16)vo;
}

// ---------------------------------------------------------------------------
// Flash attention.  One wave per 16-row q block; 8 waves/block.
// S and P·V both via v_wmma; P re-laid out C-format -> A-format through LDS.
// ---------------------------------------------------------------------------
__device__ inline float redmax16(float v) {
#pragma unroll
  for (int i = 1; i < 16; i <<= 1) v = fmaxf(v, __shfl_xor(v, i, 16));
  return v;
}
__device__ inline float redsum16(float v) {
#pragma unroll
  for (int i = 1; i < 16; i <<= 1) v += __shfl_xor(v, i, 16);
  return v;
}

__global__ void __launch_bounds__(256)
flash_attn_kernel(const _Float16* __restrict__ q, const _Float16* __restrict__ kk,
                  const _Float16* __restrict__ vt, const int* __restrict__ ymask,
                  _Float16* __restrict__ ybuf) {
  __shared__ _Float16 pshm[8][16][32];   // per-wave P tile (16 q-rows x 32 keys)
  int wave = threadIdx.x >> 5;
  int lane = threadIdx.x & 31;
  int hi   = lane >> 4;
  int l16  = lane & 15;
  int kbA  = hi * 16;
  int kbB  = hi * 16;

  int bh = blockIdx.y;
  int b  = bh >> 4;
  int h  = bh & 15;
  int t0 = (blockIdx.x * 8 + wave) * 16;

  const _Float16* qh = q  + (size_t)bh * TSEQ * HDIM;
  const _Float16* kh = kk + (size_t)bh * TSEQ * HDIM;
  const _Float16* vh = vt + (size_t)bh * HDIM * TSEQ;

  // q fragments for K(dim) = 0..31 and 32..63 (A-format), reused across key tiles
  union { v16h v; v8h h2[2]; } qa0, qa1;
  {
    const _Float16* qp = qh + (size_t)(t0 + l16) * HDIM;
    qa0.h2[0] = *(const v8h*)(qp + kbA);        qa0.h2[1] = *(const v8h*)(qp + kbA + 8);
    qa1.h2[0] = *(const v8h*)(qp + 32 + kbA);   qa1.h2[1] = *(const v8h*)(qp + 32 + kbA + 8);
  }

  // per-lane row text-mask bits (rows this lane's accumulator elements touch)
  int ymr[8];
#pragma unroll
  for (int r = 0; r < 8; ++r) {
    int row = t0 + hi * 8 + r;
    ymr[r] = (row < 64) ? ymask[b * 64 + row] : 0;
  }

  v8f yacc[4] = {};
  float mrow[8], lrow[8];
#pragma unroll
  for (int r = 0; r < 8; ++r) { mrow[r] = NEGBIG; lrow[r] = 0.f; }

  int kend = t0 + 16;
  if (kend < 64) kend = 64;          // text mask can reach keys < 64
  kend = (kend + 31) & ~31;

  const float sscale = 0.125f;       // 1/sqrt(64)

  for (int kt = 0; kt < kend; kt += 32) {
    // ---- S = Q @ K^T for a 16 x 32 key tile (two 16-col accumulators) ----
    int keyA = kt + l16;
    int keyB = kt + 16 + l16;
    union { v16h v; v8h h2[2]; } ba, bb;
    v8f sa = {}, sb = {};

    const _Float16* kpA = kh + (size_t)keyA * HDIM;
    const _Float16* kpB = kh + (size_t)keyB * HDIM;
    ba.h2[0] = *(const v8h*)(kpA + kbB);      ba.h2[1] = *(const v8h*)(kpA + kbB + 8);
    sa = __builtin_amdgcn_wmma_f32_16x16x32_f16(false, qa0.v, false, ba.v, (short)0, sa, false, false);
    ba.h2[0] = *(const v8h*)(kpA + 32 + kbB); ba.h2[1] = *(const v8h*)(kpA + 32 + kbB + 8);
    sa = __builtin_amdgcn_wmma_f32_16x16x32_f16(false, qa1.v, false, ba.v, (short)0, sa, false, false);

    bb.h2[0] = *(const v8h*)(kpB + kbB);      bb.h2[1] = *(const v8h*)(kpB + kbB + 8);
    sb = __builtin_amdgcn_wmma_f32_16x16x32_f16(false, qa0.v, false, bb.v, (short)0, sb, false, false);
    bb.h2[0] = *(const v8h*)(kpB + 32 + kbB); bb.h2[1] = *(const v8h*)(kpB + 32 + kbB + 8);
    sb = __builtin_amdgcn_wmma_f32_16x16x32_f16(false, qa1.v, false, bb.v, (short)0, sb, false, false);

    int ymA = (keyA < 64) ? ymask[b * 64 + keyA] : 0;
    int ymB = (keyB < 64) ? ymask[b * 64 + keyB] : 0;

    // ---- online softmax update per row ----
#pragma unroll
    for (int r = 0; r < 8; ++r) {
      int row = t0 + hi * 8 + r;
      bool alA = (keyA <= row) || (ymA && ymr[r]);
      bool alB = (keyB <= row) || (ymB && ymr[r]);
      float va = alA ? sa[r] * sscale : NEGBIG;
      float vb = alB ? sb[r] * sscale : NEGBIG;

      float mx = redmax16(fmaxf(va, vb));
      float mnew = fmaxf(mrow[r], mx);
      float alpha = __expf(mrow[r] - mnew);
      float pa = __expf(va - mnew);
      float pb = __expf(vb - mnew);
      float rsum = redsum16(pa + pb);
      lrow[r] = lrow[r] * alpha + rsum;
      mrow[r] = mnew;
#pragma unroll
      for (int nt = 0; nt < 4; ++nt) yacc[nt][r] *= alpha;
      pshm[wave][hi * 8 + r][l16]      = (_Float16)pa;
      pshm[wave][hi * 8 + r][16 + l16] = (_Float16)pb;
    }

    asm volatile("s_wait_dscnt 0" ::: "memory");

    // ---- P (A-format from LDS) @ V tile ----
    union { v16h v; v8h h2[2]; } pf;
    pf.h2[0] = *(const v8h*)&pshm[wave][l16][kbA];
    pf.h2[1] = *(const v8h*)&pshm[wave][l16][kbA + 8];

#pragma unroll
    for (int nt = 0; nt < 4; ++nt) {
      int d = nt * 16 + l16;
      union { v16h v; v8h h2[2]; } bv;
      const _Float16* vp = vh + (size_t)d * TSEQ + kt + kbB;
      bv.h2[0] = *(const v8h*)(vp);
      bv.h2[1] = *(const v8h*)(vp + 8);
      yacc[nt] = __builtin_amdgcn_wmma_f32_16x16x32_f16(false, pf.v, false, bv.v, (short)0, yacc[nt], false, false);
    }
  }

  // ---- normalize and write y as f16 [B,T,C] ----
#pragma unroll
  for (int nt = 0; nt < 4; ++nt) {
    int d = nt * 16 + l16;
#pragma unroll
    for (int r = 0; r < 8; ++r) {
      int row = t0 + hi * 8 + r;
      ybuf[((size_t)(b * TSEQ + row)) * CEMB + h * HDIM + d] = (_Float16)(yacc[nt][r] / lrow[r]);
    }
  }
}

// ---------------------------------------------------------------------------
// Host-side orchestration
// ---------------------------------------------------------------------------
static inline size_t align256(size_t x) { return (x + 255) & ~(size_t)255; }

extern "C" void kernel_launch(void* const* d_in, const int* in_sizes, int n_in,
                              void* d_out, int out_size, void* d_ws, size_t ws_size,
                              hipStream_t stream) {
  const float* x      = (const float*)d_in[0];
  const int*   ym     = (const int*)  d_in[1];
  const float* Wqkv   = (const float*)d_in[2];
  const float* Wattn  = (const float*)d_in[3];
  const float* scale1 = (const float*)d_in[4];
  const float* scale2 = (const float*)d_in[5];
  const float* Wfc1   = (const float*)d_in[6];
  const float* Wfc2   = (const float*)d_in[7];
  const float* Wmlp   = (const float*)d_in[8];
  float* out = (float*)d_out;

  char* ws = (char*)d_ws;
  size_t off = 0;
  auto alloc = [&](size_t bytes) { char* p = ws + off; off = align256(off + bytes); return p; };

  _Float16* wqkv_t = (_Float16*)alloc((size_t)3072 * 1024 * 2);
  _Float16* wattn_t= (_Float16*)alloc((size_t)1024 * 1024 * 2);
  _Float16* wfc1_t = (_Float16*)alloc((size_t)NHID * 1024 * 2);
  _Float16* wfc2_t = (_Float16*)alloc((size_t)NHID * 1024 * 2);
  _Float16* wmlp_t = (_Float16*)alloc((size_t)1024 * NHID * 2);
  _Float16* h1     = (_Float16*)alloc((size_t)MROWS * CEMB * 2);
  float*    qkv    = (float*)   alloc((size_t)MROWS * 3 * CEMB * 4);
  _Float16* qf     = (_Float16*)alloc((size_t)NB * NHEAD * TSEQ * HDIM * 2);
  _Float16* kf     = (_Float16*)alloc((size_t)NB * NHEAD * TSEQ * HDIM * 2);
  _Float16* vtf    = (_Float16*)alloc((size_t)NB * NHEAD * HDIM * TSEQ * 2);
  _Float16* ybuf   = (_Float16*)alloc((size_t)MROWS * CEMB * 2);
  float*    x2     = (float*)   alloc((size_t)MROWS * CEMB * 4);
  _Float16* h2     = (_Float16*)alloc((size_t)MROWS * CEMB * 2);
  float*    u1     = (float*)   alloc((size_t)MROWS * NHID * 4);
  _Float16* mbuf   = (_Float16*)alloc((size_t)MROWS * NHID * 2);
  (void)ws_size; (void)in_sizes; (void)n_in; (void)out_size;

  // 1) weights -> f16 transposed
  cast_wt_kernel<<<(1024 * 3072 + 255) / 256, 256, 0, stream>>>(Wqkv,  wqkv_t, 1024, 3072);
  cast_wt_kernel<<<(1024 * 1024 + 255) / 256, 256, 0, stream>>>(Wattn, wattn_t, 1024, 1024);
  cast_wt_kernel<<<(1024 * NHID + 255) / 256, 256, 0, stream>>>(Wfc1,  wfc1_t, 1024, NHID);
  cast_wt_kernel<<<(1024 * NHID + 255) / 256, 256, 0, stream>>>(Wfc2,  wfc2_t, 1024, NHID);
  cast_wt_kernel<<<(NHID * 1024 + 255) / 256, 256, 0, stream>>>(Wmlp,  wmlp_t, NHID, 1024);

  // 2) h1 = rmsnorm(x) * scale1  (f16)
  rmsnorm_cast_kernel<<<MROWS, 256, 0, stream>>>(x, scale1, h1, CEMB);

  // 3) qkv = h1 @ Wqkv   (f32)
  gemm_wmma_kernel<<<dim3(3072 / 64, MROWS / 128), 256, 0, stream>>>(
      h1, wqkv_t, qkv, nullptr, nullptr, MROWS, 3072, CEMB, 0);

  // 4) RoPE + pack q/k, transpose v
  {
    size_t total = (size_t)NB * NHEAD * TSEQ * (HDIM / 2);
    rope_pack_kernel<<<(unsigned)((total + 255) / 256), 256, 0, stream>>>(qkv, qf, kf, vtf);
  }

  // 5) attention (flash, masked softmax) -> ybuf f16 [B,T,C]
  flash_attn_kernel<<<dim3(TSEQ / 128, NB * NHEAD), 256, 0, stream>>>(qf, kf, vtf, ym, ybuf);

  // 6) x2 = x + ybuf @ Wattn_proj
  gemm_wmma_kernel<<<dim3(CEMB / 64, MROWS / 128), 256, 0, stream>>>(
      ybuf, wattn_t, x2, nullptr, x, MROWS, CEMB, CEMB, 1);

  // 7) h2 = rmsnorm(x2) * scale2
  rmsnorm_cast_kernel<<<MROWS, 256, 0, stream>>>(x2, scale2, h2, CEMB);

  // 8) u1 = h2 @ Wfc1
  gemm_wmma_kernel<<<dim3(NHID / 64, MROWS / 128), 256, 0, stream>>>(
      h2, wfc1_t, u1, nullptr, nullptr, MROWS, NHID, CEMB, 0);

  // 9) mbuf = silu(u1) * (h2 @ Wfc2)   (f16)
  gemm_wmma_kernel<<<dim3(NHID / 64, MROWS / 128), 256, 0, stream>>>(
      h2, wfc2_t, nullptr, mbuf, u1, MROWS, NHID, CEMB, 2);

  // 10) out = x2 + mbuf @ Wmlp_proj
  gemm_wmma_kernel<<<dim3(CEMB / 64, MROWS / 128), 256, 0, stream>>>(
      mbuf, wmlp_t, out, nullptr, x2, MROWS, CEMB, NHID, 1);
}
